// QLSTM_65481071397453
// MI455X (gfx1250) — matrix operand order repaired and checked by reading, not compiled, not run on hardware
//
#include <hip/hip_runtime.h>
#include <hip/hip_bf16.h>

// QLSTM fused scan for MI455X (gfx1250, wave32).
//
// Design (see analysis): batch rows are independent through the recurrence, so
// each wave owns a 16-row batch tile and runs the whole 512-step scan with
// h/c state in registers. Matmuls are v_wmma_f32_16x16x32_f16 with
// M=hidden(16), N=batch-tile(16), K=fan (80 padded to 96 for layer 0, 32 for
// layer 1). The WMMA C-layout (lane holds one batch column, 8 hidden rows)
// makes the qlayer cumprod an in-lane scan + one cross-half __shfl, and makes
// the C->B relayout of h for the next matmul a single __shfl per register.

typedef __attribute__((ext_vector_type(16))) _Float16 v16h;
typedef __attribute__((ext_vector_type(8)))  float    v8f;

#define SEQ_LEN_ 512
#define BATCH_   1024
#define IN_DIM_  64
#define HID_     16

struct QParams {
    const float* x;        // [T, B, IN]
    const float* W[2][4];  // [16, fan] row-major, fan = 80 (l0) / 32 (l1)
    const float* b[2][4];  // [16]
    const float* t[2][4];  // [16]
    float* out;            // [T*B*16 | h_l0 | h_l1 | c_l0 | c_l1]
};

static __device__ __forceinline__ v8f wmma_f16(v16h a, v16h b, v8f c) {
    // (neg_a, A, neg_b, B, c_mod, C, reuse_a, reuse_b)
    return __builtin_amdgcn_wmma_f32_16x16x32_f16(
        false, a, false, b, (short)0, c, false, false);
}

// h is in WMMA C-layout: lane (hi,col) holds rows n = r + 8*hi, column b=col.
// Produce seq[j] = h[n=j] for this lane's column (needs the other half's 8
// values via one cross-half shuffle per register).
static __device__ __forceinline__ void col_seq(const float h[8], int lane, int hi,
                                               float seq[16]) {
#pragma unroll
    for (int r = 0; r < 8; ++r) {
        float oth = __shfl(h[r], lane ^ 16, 32);
        seq[r]     = hi ? oth  : h[r];
        seq[r + 8] = hi ? h[r] : oth;
    }
}

// Build a 32x16 f16 B fragment. Lane-half selects K range: lower lanes carry
// K=0..15 (source lo), upper lanes K=16..31 (source hiSeq). N = lane%16.
static __device__ __forceinline__ v16h make_b(const float lo[16],
                                              const float hiSeq[16], int hi) {
    v16h b;
#pragma unroll
    for (int e = 0; e < 16; ++e)
        b[e] = (_Float16)(hi ? hiSeq[e] : lo[e]);
    return b;
}

// qlayer + activation: out[n] = act( cumprod_{j<=n} cos(z_j + (b_j + th_j)) ).
// acc holds z for rows n = r + 8*hi; cumprod over n = in-lane scan + one
// cross-half shuffle of the lower half's total product.
template <bool TANH>
static __device__ __forceinline__ void qgate(const v8f& acc, const float bt[8],
                                             int lane, int hi, float out[8]) {
    float cum[8];
    float prod = 1.0f;
#pragma unroll
    for (int r = 0; r < 8; ++r) {
        prod *= __cosf(acc[r] + bt[r]);
        cum[r] = prod;
    }
    float lowTot = __shfl(prod, lane & 15, 32);  // lower lane's prod(n=0..7)
    float scale  = hi ? lowTot : 1.0f;
#pragma unroll
    for (int r = 0; r < 8; ++r) {
        float z = cum[r] * scale;
        if (TANH) {
            float e2 = __expf(-2.0f * z);
            out[r] = (1.0f - e2) / (1.0f + e2);
        } else {
            out[r] = 1.0f / (1.0f + __expf(-z));
        }
    }
}

static __device__ __forceinline__ v16h cvt16(float4 a, float4 b, float4 c, float4 d) {
    v16h v;
    v[0]=(_Float16)a.x;  v[1]=(_Float16)a.y;  v[2]=(_Float16)a.z;  v[3]=(_Float16)a.w;
    v[4]=(_Float16)b.x;  v[5]=(_Float16)b.y;  v[6]=(_Float16)b.z;  v[7]=(_Float16)b.w;
    v[8]=(_Float16)c.x;  v[9]=(_Float16)c.y;  v[10]=(_Float16)c.z; v[11]=(_Float16)c.w;
    v[12]=(_Float16)d.x; v[13]=(_Float16)d.y; v[14]=(_Float16)d.z; v[15]=(_Float16)d.w;
    return v;
}

__global__ __launch_bounds__(32) void qlstm_fused(QParams p) {
    const int lane = threadIdx.x & 31;
    const int hi   = lane >> 4;   // lane half
    const int col  = lane & 15;   // batch column within tile / weight row M
    const int bcol = blockIdx.x * 16 + col;

    // ---- Preload weights as WMMA A fragments (M=hidden row = col) ----------
    // A operand, contiguous K order used here (same as B): lane half hi holds K = {16*hi+e : e<16}
    // of row M = lane%16.
    v16h A0[4][3];           // layer 0: K chunks 0..2 (K=64..79 is h, 80..95 pad 0)
    v16h A1[4];              // layer 1: K=32 exactly
    float bt0[4][8], bt1[4][8];  // bias + theta folded (cos(z + b + th))
#pragma unroll
    for (int g = 0; g < 4; ++g) {
        const float* W0 = p.W[0][g];  // [16 x 80]
#pragma unroll
        for (int c = 0; c < 3; ++c) {
            v16h a;
#pragma unroll
            for (int e = 0; e < 16; ++e) {
                int k = 32 * c + 16 * hi + e;
                a[e] = (_Float16)((k < 80) ? W0[col * 80 + k] : 0.0f);
            }
            A0[g][c] = a;
        }
        const float* W1 = p.W[1][g];  // [16 x 32]
        v16h a1;
#pragma unroll
        for (int e = 0; e < 16; ++e) {
            int k = 16 * hi + e;
            a1[e] = (_Float16)W1[col * 32 + k];
        }
        A1[g] = a1;
#pragma unroll
        for (int r = 0; r < 8; ++r) {
            int n = r + 8 * hi;
            bt0[g][r] = p.b[0][g][n] + p.t[0][g][n];
            bt1[g][r] = p.b[1][g][n] + p.t[1][g][n];
        }
    }

    // ---- Recurrent state in registers (C-layout) ---------------------------
    float h0[8], c0[8], h1[8], c1[8];
    float zseq[16];
#pragma unroll
    for (int r = 0; r < 8; ++r) { h0[r] = 0.f; c0[r] = 0.f; h1[r] = 0.f; c1[r] = 0.f; }
#pragma unroll
    for (int e = 0; e < 16; ++e) zseq[e] = 0.f;

    for (int tt = 0; tt < SEQ_LEN_; ++tt) {
        // B fragments for x: lane reads 16 contiguous f32 (64B aligned).
        const float* xp = p.x + ((size_t)tt * BATCH_ + bcol) * IN_DIM_ + 16 * hi;
        const float4* xv = (const float4*)xp;
        v16h Bx0 = cvt16(xv[0], xv[1], xv[2],  xv[3]);   // K = 0..31
        v16h Bx1 = cvt16(xv[8], xv[9], xv[10], xv[11]);  // K = 32..63

        // h0 occupies K=64..79 of chunk 2; upper half (K=80..95) is zero pad.
        float s0[16];
        col_seq(h0, lane, hi, s0);
        v16h Bh = make_b(s0, zseq, hi);

        // ---- Layer 0 gates -------------------------------------------------
        float fg[8], ig[8], gg[8], og[8];
        {
            v8f acc = {};
            acc = wmma_f16(A0[0][0], Bx0, acc);
            acc = wmma_f16(A0[0][1], Bx1, acc);
            acc = wmma_f16(A0[0][2], Bh,  acc);
            qgate<false>(acc, bt0[0], lane, hi, fg);
        }
        {
            v8f acc = {};
            acc = wmma_f16(A0[1][0], Bx0, acc);
            acc = wmma_f16(A0[1][1], Bx1, acc);
            acc = wmma_f16(A0[1][2], Bh,  acc);
            qgate<false>(acc, bt0[1], lane, hi, ig);
        }
        {
            v8f acc = {};
            acc = wmma_f16(A0[2][0], Bx0, acc);
            acc = wmma_f16(A0[2][1], Bx1, acc);
            acc = wmma_f16(A0[2][2], Bh,  acc);
            qgate<true>(acc, bt0[2], lane, hi, gg);
        }
        {
            v8f acc = {};
            acc = wmma_f16(A0[3][0], Bx0, acc);
            acc = wmma_f16(A0[3][1], Bx1, acc);
            acc = wmma_f16(A0[3][2], Bh,  acc);
            qgate<false>(acc, bt0[3], lane, hi, og);
        }
        float h0n[8];
#pragma unroll
        for (int r = 0; r < 8; ++r) {
            float cc = fg[r] * c0[r] + ig[r] * gg[r];
            c0[r] = cc;
            float e2 = __expf(-2.0f * cc);
            h0n[r] = og[r] * (1.0f - e2) / (1.0f + e2);
        }

        // ---- Layer 1: comb = [h0_new (K=0..15) | h1 (K=16..31)] ------------
        float sl[16], sh[16];
        col_seq(h0n, lane, hi, sl);
        col_seq(h1, lane, hi, sh);
        v16h B1 = make_b(sl, sh, hi);
        {
            v8f acc = {};
            acc = wmma_f16(A1[0], B1, acc);
            qgate<false>(acc, bt1[0], lane, hi, fg);
        }
        {
            v8f acc = {};
            acc = wmma_f16(A1[1], B1, acc);
            qgate<false>(acc, bt1[1], lane, hi, ig);
        }
        {
            v8f acc = {};
            acc = wmma_f16(A1[2], B1, acc);
            qgate<true>(acc, bt1[2], lane, hi, gg);
        }
        {
            v8f acc = {};
            acc = wmma_f16(A1[3], B1, acc);
            qgate<false>(acc, bt1[3], lane, hi, og);
        }
#pragma unroll
        for (int r = 0; r < 8; ++r) {
            float cc = fg[r] * c1[r] + ig[r] * gg[r];
            c1[r] = cc;
            float e2 = __expf(-2.0f * cc);
            h1[r] = og[r] * (1.0f - e2) / (1.0f + e2);
            h0[r] = h0n[r];
        }

        // outputs[t, b, n]: lane's 8 hidden values are contiguous (n=8*hi+r).
        float* op = p.out + ((size_t)tt * BATCH_ + bcol) * HID_ + 8 * hi;
        *(float4*)(op)     = make_float4(h1[0], h1[1], h1[2], h1[3]);
        *(float4*)(op + 4) = make_float4(h1[4], h1[5], h1[6], h1[7]);
    }

    // ---- Final states: [h_l0 | h_l1 | c_l0 | c_l1] after outputs -----------
    const size_t base = (size_t)SEQ_LEN_ * BATCH_ * HID_;
    const size_t off  = (size_t)bcol * HID_ + 8 * hi;
    const size_t BH   = (size_t)BATCH_ * HID_;
    float* hp0 = p.out + base + off;
    float* hp1 = p.out + base + BH + off;
    float* cp0 = p.out + base + 2 * BH + off;
    float* cp1 = p.out + base + 3 * BH + off;
    *(float4*)(hp0)     = make_float4(h0[0], h0[1], h0[2], h0[3]);
    *(float4*)(hp0 + 4) = make_float4(h0[4], h0[5], h0[6], h0[7]);
    *(float4*)(hp1)     = make_float4(h1[0], h1[1], h1[2], h1[3]);
    *(float4*)(hp1 + 4) = make_float4(h1[4], h1[5], h1[6], h1[7]);
    *(float4*)(cp0)     = make_float4(c0[0], c0[1], c0[2], c0[3]);
    *(float4*)(cp0 + 4) = make_float4(c0[4], c0[5], c0[6], c0[7]);
    *(float4*)(cp1)     = make_float4(c1[0], c1[1], c1[2], c1[3]);
    *(float4*)(cp1 + 4) = make_float4(c1[4], c1[5], c1[6], c1[7]);
}

extern "C" void kernel_launch(void* const* d_in, const int* in_sizes, int n_in,
                              void* d_out, int out_size, void* d_ws, size_t ws_size,
                              hipStream_t stream) {
    (void)in_sizes; (void)n_in; (void)d_ws; (void)ws_size; (void)out_size;
    // d_in order assumed = setup_inputs() insertion order:
    //   [0] inputs, then per layer: Wf,bf,tf, Wi,bi,ti, Wg,bg,tg, Wo,bo,to.
    QParams p;
    p.x = (const float*)d_in[0];
    int idx = 1;
    for (int l = 0; l < 2; ++l) {
        for (int g = 0; g < 4; ++g) {
            p.W[l][g] = (const float*)d_in[idx++];
            p.b[l][g] = (const float*)d_in[idx++];
            p.t[l][g] = (const float*)d_in[idx++];
        }
    }
    p.out = (float*)d_out;
    qlstm_fused<<<BATCH_ / 16, 32, 0, stream>>>(p);
}
